// MyModel_87522843559995
// MI455X (gfx1250) — compile-verified, matrix operand order checked by reading, not run on hardware
//
#include <hip/hip_runtime.h>
#include <math.h>

#define HID   128
#define BB    8
#define SS    512
#define BSZ   (BB*SS)        // 4096 rows
#define G4    (4*HID)        // 512 gate width
#define NVOC  32000

typedef __bf16 bf16;
typedef __attribute__((ext_vector_type(16))) __bf16 v16bf;
typedef __attribute__((ext_vector_type(8)))  float  v8f;
typedef __attribute__((ext_vector_type(4)))  float  vf4;

// ---------------------------------------------------------------------------
// WMMA helpers (CDNA5 wave32, V_WMMA_F32_16X16X32_BF16)
// A fragment (16x32 bf16): lane m = lane&15; K = {8h..8h+7} U {16+8h..16+8h+7}
// where h = lane>>4. Two 16-byte loads per lane from a K-major tile.
// B is fed as a K-major (transposed) tile with the same per-lane striping.
// C/D: element r of v8f -> row M = r + 8h, col N = lane&15.
// ---------------------------------------------------------------------------
__device__ __forceinline__ v16bf load_frag(const bf16* tile, int ldr) {
  int lane = threadIdx.x & 31;
  int half = lane >> 4;
  const bf16* row = tile + (size_t)(lane & 15) * ldr;
  vf4 lo = *(const vf4*)(row + 8 * half);
  vf4 hi = *(const vf4*)(row + 16 + 8 * half);
  v16bf v;
  *((vf4*)&v)     = lo;
  *((vf4*)&v + 1) = hi;
  return v;
}

__device__ __forceinline__ v8f wmma_bf16(v16bf a, v16bf b, v8f c) {
  return __builtin_amdgcn_wmma_f32_16x16x32_bf16(false, a, false, b,
                                                 (short)0, c, false, false);
}

__device__ __forceinline__ float sigmoidf_(float x) {
  return 1.0f / (1.0f + expf(-x));
}

// ---------------------------------------------------------------------------
// Embedding gather: X[bs, h] = emb[tokens[bs], h]  (fp32 -> bf16)
// ---------------------------------------------------------------------------
__global__ __launch_bounds__(256) void embed_kernel(const int* __restrict__ tokens,
                                                    const float* __restrict__ emb,
                                                    bf16* __restrict__ X) {
  int idx = blockIdx.x * 256 + threadIdx.x;     // 0 .. BSZ*HID-1
  int bs = idx >> 7;
  int h  = idx & 127;
  int tok = tokens[bs];
  X[idx] = (bf16)emb[(size_t)tok * HID + h];
}

// ---------------------------------------------------------------------------
// Tiled transpose + convert: Bt[n][k] = (bf16) B[k][n]
// B: [K][N] fp32 row-major, Bt: [N][K] bf16.  K, N multiples of 32.
// Grid: (N/32, K/32), block (32, 8). Coalesced both directions via LDS.
// ---------------------------------------------------------------------------
__global__ __launch_bounds__(256) void transcvt_kernel(const float* __restrict__ B,
                                                       bf16* __restrict__ Bt,
                                                       int K, int N) {
  __shared__ bf16 tile[32][33];   // [k][n], padded to dodge bank conflicts
  int n0 = blockIdx.x * 32;
  int k0 = blockIdx.y * 32;
  int tx = threadIdx.x;           // 0..31
  int ty = threadIdx.y;           // 0..7
#pragma unroll
  for (int i = 0; i < 32; i += 8)
    tile[ty + i][tx] = (bf16)B[(size_t)(k0 + ty + i) * N + n0 + tx];
  __syncthreads();
#pragma unroll
  for (int i = 0; i < 32; i += 8)
    Bt[(size_t)(n0 + ty + i) * K + k0 + tx] = tile[tx][ty + i];
}

// ---------------------------------------------------------------------------
// GEMM (TN, all-bf16 operands, f32 accumulate, K fixed = HID = 128):
//   C[M,N] = A(bf16, [M][128] K-major) @ Bt(bf16, [N][128] K-major)^T + bias
// Block = 256 thr = 8 waves arranged 2(M) x 4(N); each wave owns a 2x2 grid
// of 16x16 accumulators -> block tile 64(M) x 128(N). Fragments are loaded
// directly from global (A: 1MB, Bt: <=8MB -> L2-resident), no LDS staging.
// M % 64 == 0, N % 128 == 0 (holds: 4096, {512, 32000}).
// ---------------------------------------------------------------------------
__global__ __launch_bounds__(256) void gemm_tn_kernel(const bf16* __restrict__ A,
                                                      const bf16* __restrict__ Bt,
                                                      const float* __restrict__ bias,
                                                      float* __restrict__ C,
                                                      int M, int N) {
  int tid  = threadIdx.x;
  int wave = tid >> 5;
  int lane = tid & 31;
  int wm = wave >> 2;                    // 0..1
  int wn = wave & 3;                     // 0..3
  int m0 = blockIdx.y * 64 + wm * 32;
  int n0 = blockIdx.x * 128 + wn * 32;

  v8f acc00 = {}, acc01 = {}, acc10 = {}, acc11 = {};
  const bf16* Arow0 = A  + (size_t)m0 * HID;
  const bf16* Arow1 = A  + (size_t)(m0 + 16) * HID;
  const bf16* Brow0 = Bt + (size_t)n0 * HID;
  const bf16* Brow1 = Bt + (size_t)(n0 + 16) * HID;

#pragma unroll
  for (int k0 = 0; k0 < HID; k0 += 32) {
    v16bf a0 = load_frag(Arow0 + k0, HID);
    v16bf a1 = load_frag(Arow1 + k0, HID);
    v16bf b0 = load_frag(Brow0 + k0, HID);
    v16bf b1 = load_frag(Brow1 + k0, HID);
    acc00 = wmma_bf16(a0, b0, acc00);
    acc01 = wmma_bf16(a0, b1, acc01);
    acc10 = wmma_bf16(a1, b0, acc10);
    acc11 = wmma_bf16(a1, b1, acc11);
  }

  int half = lane >> 4;
  int nn   = lane & 15;
  float bv0 = bias[n0 + nn];
  float bv1 = bias[n0 + 16 + nn];
#pragma unroll
  for (int r = 0; r < 8; ++r) {
    int row0 = m0 + half * 8 + r;
    int row1 = row0 + 16;
    C[(size_t)row0 * N + n0 + nn]      = acc00[r] + bv0;
    C[(size_t)row0 * N + n0 + 16 + nn] = acc01[r] + bv1;
    C[(size_t)row1 * N + n0 + nn]      = acc10[r] + bv0;
    C[(size_t)row1 * N + n0 + 16 + nn] = acc11[r] + bv1;
  }
}

// ---------------------------------------------------------------------------
// Masked LSTM scan (one persistent workgroup, 512 thr = 16 waves).
// Z[bs, 0:512] = x_t@W + b precomputed. Per step: zr = h_pad(16x128) @ U,
// gates, masked state update. Ut is [G4][HID] bf16 (L2-resident).
// Hout[bs, j] = h after update (bf16, feeds next GEMM).
// ---------------------------------------------------------------------------
__global__ __launch_bounds__(512) void lstm_scan_kernel(const float* __restrict__ Z,
                                                        const bf16* __restrict__ Ut,
                                                        const int* __restrict__ tokens,
                                                        bf16* __restrict__ Hout) {
  __shared__ alignas(16) bf16  hA[16][128];   // padded h in WMMA A layout (bf16)
  __shared__ float hf[8][128];                // fp32 h state
  __shared__ float cf[8][128];                // fp32 c state
  __shared__ alignas(16) float zrS[16][512];  // recurrent gate contributions

  int tid  = threadIdx.x;
  int wave = tid >> 5;
  int lane = tid & 31;

  for (int i = tid; i < 16 * 128; i += 512) (&hA[0][0])[i] = (bf16)0.0f;
  for (int i = tid; i < 8 * 128; i += 512) {
    (&hf[0][0])[i] = 0.0f;
    (&cf[0][0])[i] = 0.0f;
  }
  __syncthreads();

  int j  = tid & 127;   // hidden index
  int bq = tid >> 7;    // 0..3 -> batches bq and bq+4

  for (int t = 0; t < SS; ++t) {
    // phase 1: zr = hA @ Ut  (32 N-tiles of 16; 2 per wave; K = 128 = 4x32)
#pragma unroll
    for (int q = 0; q < 2; ++q) {
      int ntile = wave + q * 16;
      v8f acc = {};
#pragma unroll
      for (int kk = 0; kk < 4; ++kk) {
        v16bf a = load_frag(&hA[0][kk * 32], 128);
        v16bf b = load_frag(Ut + (size_t)(ntile * 16) * 128 + kk * 32, 128);
        acc = wmma_bf16(a, b, acc);
      }
      int half = lane >> 4;
      int n    = lane & 15;
#pragma unroll
      for (int r = 0; r < 8; ++r)
        zrS[half * 8 + r][ntile * 16 + n] = acc[r];
    }
    __syncthreads();

    // phase 2: gates + masked update for b = bq and bq+4
#pragma unroll
    for (int u = 0; u < 2; ++u) {
      int b  = bq + u * 4;
      int bs = (b << 9) + t;
      const float* zrow = Z + (size_t)bs * G4;
      float zi = zrow[j]           + zrS[b][j];
      float zf = zrow[128 + j]     + zrS[b][128 + j];
      float zg = zrow[256 + j]     + zrS[b][256 + j];
      float zo = zrow[384 + j]     + zrS[b][384 + j];
      float ig = sigmoidf_(zi);
      float fg = sigmoidf_(zf);
      float gg = tanhf(zg);
      float og = sigmoidf_(zo);
      float c  = cf[b][j];
      float cn = fg * c + ig * gg;
      float h  = hf[b][j];
      float hn = og * tanhf(cn);
      if (tokens[bs] != 0) { h = hn; c = cn; }
      hf[b][j] = h;
      cf[b][j] = c;
      hA[b][j] = (bf16)h;                       // feeds next step's WMMA
      Hout[(size_t)bs * HID + j] = (bf16)h;     // ys output
    }
    __syncthreads();
  }
}

// ---------------------------------------------------------------------------
// Row softmax: probs[row,:] = softmax(logits[row,:]), N = 32000, 1 block/row
// ---------------------------------------------------------------------------
__global__ __launch_bounds__(256) void softmax_kernel(const float* __restrict__ logits,
                                                      float* __restrict__ probs,
                                                      int N) {
  __shared__ float red[256];
  int row = blockIdx.x;
  const float* x = logits + (size_t)row * N;
  float*       p = probs  + (size_t)row * N;

  float m = -3.4e38f;
  for (int i = threadIdx.x; i < N; i += 256) m = fmaxf(m, x[i]);
  red[threadIdx.x] = m;
  __syncthreads();
  for (int s = 128; s > 0; s >>= 1) {
    if (threadIdx.x < s) red[threadIdx.x] = fmaxf(red[threadIdx.x], red[threadIdx.x + s]);
    __syncthreads();
  }
  m = red[0];
  __syncthreads();

  float sum = 0.0f;
  for (int i = threadIdx.x; i < N; i += 256) sum += expf(x[i] - m);
  red[threadIdx.x] = sum;
  __syncthreads();
  for (int s = 128; s > 0; s >>= 1) {
    if (threadIdx.x < s) red[threadIdx.x] += red[threadIdx.x + s];
    __syncthreads();
  }
  float inv = 1.0f / red[0];
  __syncthreads();

  for (int i = threadIdx.x; i < N; i += 256) p[i] = expf(x[i] - m) * inv;
}

// ---------------------------------------------------------------------------
extern "C" void kernel_launch(void* const* d_in, const int* in_sizes, int n_in,
                              void* d_out, int out_size, void* d_ws, size_t ws_size,
                              hipStream_t stream) {
  const int*   tokens = (const int*)  d_in[0];
  const float* emb    = (const float*)d_in[1];
  const float* W0     = (const float*)d_in[2];
  const float* U0     = (const float*)d_in[3];
  const float* b0     = (const float*)d_in[4];
  const float* W1     = (const float*)d_in[5];
  const float* U1     = (const float*)d_in[6];
  const float* b1     = (const float*)d_in[7];
  const float* Wd     = (const float*)d_in[8];
  const float* bd     = (const float*)d_in[9];

  float* logits = (float*)d_out;
  float* probs  = logits + (size_t)BSZ * NVOC;

  char* ws = (char*)d_ws;
  const size_t MB = 1u << 20;
  bf16*  Xbf = (bf16*) (ws);                         // 1 MB   [4096][128]
  float* Z   = (float*)(ws + 1  * MB);               // 8 MB   [4096][512]
  bf16*  H1  = (bf16*) (ws + 9  * MB);               // 1 MB
  bf16*  H2  = (bf16*) (ws + 10 * MB);               // 1 MB
  bf16*  Ut0 = (bf16*) (ws + 11 * MB);               // 128 KB [512][128]
  bf16*  Ut1 = (bf16*) (ws + 11 * MB + (1u << 17));  // 128 KB
  bf16*  Wt0 = (bf16*) (ws + 11 * MB + (2u << 17));  // 128 KB [512][128]
  bf16*  Wt1 = (bf16*) (ws + 11 * MB + (3u << 17));  // 128 KB
  bf16*  Wdt = (bf16*) (ws + 12 * MB);               // 8 MB   [32000][128]

  embed_kernel<<<(BSZ * HID) / 256, 256, 0, stream>>>(tokens, emb, Xbf);

  dim3 tcvtThr(32, 8);
  transcvt_kernel<<<dim3(G4 / 32,   HID / 32), tcvtThr, 0, stream>>>(U0, Ut0, HID, G4);
  transcvt_kernel<<<dim3(G4 / 32,   HID / 32), tcvtThr, 0, stream>>>(U1, Ut1, HID, G4);
  transcvt_kernel<<<dim3(G4 / 32,   HID / 32), tcvtThr, 0, stream>>>(W0, Wt0, HID, G4);
  transcvt_kernel<<<dim3(G4 / 32,   HID / 32), tcvtThr, 0, stream>>>(W1, Wt1, HID, G4);
  transcvt_kernel<<<dim3(NVOC / 32, HID / 32), tcvtThr, 0, stream>>>(Wd, Wdt, HID, NVOC);

  gemm_tn_kernel<<<dim3(G4 / 128, BSZ / 64), 256, 0, stream>>>(Xbf, Wt0, b0, Z, BSZ, G4);
  lstm_scan_kernel<<<1, 512, 0, stream>>>(Z, Ut0, tokens, H1);

  gemm_tn_kernel<<<dim3(G4 / 128, BSZ / 64), 256, 0, stream>>>(H1, Wt1, b1, Z, BSZ, G4);
  lstm_scan_kernel<<<1, 512, 0, stream>>>(Z, Ut1, tokens, H2);

  gemm_tn_kernel<<<dim3(NVOC / 128, BSZ / 64), 256, 0, stream>>>(H2, Wdt, bd, logits, BSZ, NVOC);
  softmax_kernel<<<BSZ, 256, 0, stream>>>(logits, probs, NVOC);
}
